// TromptCell_28235115004155
// MI455X (gfx1250) — compile-verified, matrix operand order checked by reading, not run on hardware
//
#include <hip/hip_runtime.h>
#include <hip/hip_bf16.h>

typedef __attribute__((ext_vector_type(8)))  _Float16 v8h;
typedef __attribute__((ext_vector_type(16))) _Float16 v16h;
typedef __attribute__((ext_vector_type(8)))  float    v8f;

typedef int v4i __attribute__((vector_size(16)));
typedef __attribute__((address_space(1))) v4i gv4i;  // global 128-bit
typedef __attribute__((address_space(3))) v4i lv4i;  // LDS    128-bit

#define LN_EPS 1e-5f

#if defined(__has_builtin)
#if __has_builtin(__builtin_amdgcn_global_load_async_to_lds_b128)
#define HAVE_ASYNC_LDS 1
#endif
#endif

// 16B global -> LDS staging (async CDNA5 path when available)
__device__ __forceinline__ void stage16(const _Float16* g, _Float16* l) {
#ifdef HAVE_ASYNC_LDS
    __builtin_amdgcn_global_load_async_to_lds_b128(
        (gv4i*)(uintptr_t)g, (lv4i*)(uintptr_t)l, 0, 0);
#else
    *reinterpret_cast<v8h*>(l) = *reinterpret_cast<const v8h*>(g);
#endif
}

__device__ __forceinline__ void stage_wait() {
#ifdef HAVE_ASYNC_LDS
#if __has_builtin(__builtin_amdgcn_s_wait_asynccnt)
    __builtin_amdgcn_s_wait_asynccnt(0);
#else
    asm volatile("s_wait_asynccnt 0" ::: "memory");
#endif
#endif
}

// ---------------- block reduction helpers (blockDim.x == 256, 8 waves) ----------------
__device__ __forceinline__ float block_reduce_sum(float v, float* scratch) {
#pragma unroll
    for (int o = 16; o > 0; o >>= 1) v += __shfl_xor(v, o, 32);
    if ((threadIdx.x & 31) == 0) scratch[threadIdx.x >> 5] = v;
    __syncthreads();
    float r = 0.0f;
#pragma unroll
    for (int i = 0; i < 8; ++i) r += scratch[i];
    __syncthreads();
    return r;
}

__device__ __forceinline__ float block_reduce_max(float v, float* scratch) {
#pragma unroll
    for (int o = 16; o > 0; o >>= 1) v = fmaxf(v, __shfl_xor(v, o, 32));
    if ((threadIdx.x & 31) == 0) scratch[threadIdx.x >> 5] = v;
    __syncthreads();
    float r = scratch[0];
#pragma unroll
    for (int i = 1; i < 8; ++i) r = fmaxf(r, scratch[i]);
    __syncthreads();
    return r;
}

// ---------------- kA: x_prompt[p,d] = (concat(LN(emb_prompt[p]), prev[p]) @ W^T)[d] + b[d] + emb_prompt[p,d]
__global__ void kA_prompt(const float* __restrict__ emb_prompt,
                          const float* __restrict__ prev_cell,
                          const float* __restrict__ lnw, const float* __restrict__ lnb,
                          const float* __restrict__ W,   // [256, 512]
                          const float* __restrict__ Wb,
                          float* __restrict__ x_prompt) {
    __shared__ __align__(16) float xp[512];
    __shared__ float scratch[8];
    const int p = blockIdx.x, d = threadIdx.x;

    const float e  = emb_prompt[p * 256 + d];
    const float mu = block_reduce_sum(e, scratch) * (1.0f / 256.0f);
    const float dv = e - mu;
    const float var = block_reduce_sum(dv * dv, scratch) * (1.0f / 256.0f);
    const float rs  = rsqrtf(var + LN_EPS);
    xp[d]       = dv * rs * lnw[d] + lnb[d];
    xp[256 + d] = prev_cell[p * 256 + d];
    __syncthreads();

    float acc = Wb[d] + e;
    const float4* Wr = reinterpret_cast<const float4*>(W + (size_t)d * 512);
#pragma unroll 4
    for (int k4 = 0; k4 < 128; ++k4) {
        const float4 w4 = Wr[k4];
        const float4 x4 = *reinterpret_cast<const float4*>(&xp[k4 * 4]);
        acc += w4.x * x4.x + w4.y * x4.y + w4.z * x4.z + w4.w * x4.w;
    }
    x_prompt[p * 256 + d] = acc;
}

// ---------------- kB: x_col = LN(emb_column)
__global__ void kB_col(const float* __restrict__ emb_col,
                       const float* __restrict__ lnw, const float* __restrict__ lnb,
                       float* __restrict__ x_col) {
    __shared__ float scratch[8];
    const int c = blockIdx.x, d = threadIdx.x;
    const float e  = emb_col[c * 256 + d];
    const float mu = block_reduce_sum(e, scratch) * (1.0f / 256.0f);
    const float dv = e - mu;
    const float var = block_reduce_sum(dv * dv, scratch) * (1.0f / 256.0f);
    x_col[c * 256 + d] = dv * rsqrtf(var + LN_EPS) * lnw[d] + lnb[d];
}

// ---------------- kC: mask = softmax(x_prompt @ x_col^T)
__global__ void kC_mask(const float* __restrict__ x_prompt,
                        const float* __restrict__ x_col,
                        float* __restrict__ maskf,
                        _Float16* __restrict__ maskh) {
    __shared__ __align__(16) float xp[256];
    __shared__ float scratch[8];
    const int p = blockIdx.x, c = threadIdx.x;
    xp[c] = x_prompt[p * 256 + c];
    __syncthreads();

    float acc = 0.0f;
    const float4* xc = reinterpret_cast<const float4*>(x_col + (size_t)c * 256);
#pragma unroll 4
    for (int k4 = 0; k4 < 64; ++k4) {
        const float4 a = xc[k4];
        const float4 b = *reinterpret_cast<const float4*>(&xp[k4 * 4]);
        acc += a.x * b.x + a.y * b.y + a.z * b.z + a.w * b.w;
    }
    const float mx = block_reduce_max(acc, scratch);
    const float ex = __expf(acc - mx);
    const float sm = block_reduce_sum(ex, scratch);
    const float mv = ex / sm;
    maskf[p * 256 + c] = mv;
    maskh[p * 256 + c] = (_Float16)mv;
}

// ---------------- kD: xembT[b, d, c] = LN_over_d(relu(feb[c,:] + x[b,c]*few[c,:])) stored f16 TRANSPOSED
// grid 1024: b = bx>>2, c-tile of 64 = (bx&3)*64. block 256 (8 waves); each wave handles a pair of c-rows.
#define KD_PITCH 72
__global__ void kD_emb(const float* __restrict__ x,
                       const float* __restrict__ few, const float* __restrict__ feb,
                       const float* __restrict__ lnw, const float* __restrict__ lnb,
                       _Float16* __restrict__ xembT) {
    __shared__ __align__(16) _Float16 sT[256][KD_PITCH];  // [d][c_local]
    const int lane = threadIdx.x & 31, wave = threadIdx.x >> 5;
    const int b = blockIdx.x >> 2, c0 = (blockIdx.x & 3) << 6;

    float g[8], h[8];
#pragma unroll
    for (int i = 0; i < 8; ++i) {
        const int d = lane + 32 * i;
        g[i] = lnw[d]; h[i] = lnb[d];
    }

    for (int iter = 0; iter < 4; ++iter) {
        const int cl = iter * 16 + wave * 2;       // two adjacent c rows
        const int c  = c0 + cl;
        const float xv0 = x[b * 256 + c];
        const float xv1 = x[b * 256 + c + 1];
        float v0[8], v1[8];
        float s0 = 0.f, q0 = 0.f, s1 = 0.f, q1 = 0.f;
#pragma unroll
        for (int i = 0; i < 8; ++i) {
            const int d = lane + 32 * i;
            const float a0 = fmaxf(0.0f, feb[(size_t)c * 256 + d]       + xv0 * few[(size_t)c * 256 + d]);
            const float a1 = fmaxf(0.0f, feb[(size_t)(c + 1) * 256 + d] + xv1 * few[(size_t)(c + 1) * 256 + d]);
            v0[i] = a0; v1[i] = a1;
            s0 += a0; q0 += a0 * a0; s1 += a1; q1 += a1 * a1;
        }
#pragma unroll
        for (int o = 16; o > 0; o >>= 1) {
            s0 += __shfl_xor(s0, o, 32); q0 += __shfl_xor(q0, o, 32);
            s1 += __shfl_xor(s1, o, 32); q1 += __shfl_xor(q1, o, 32);
        }
        const float mu0 = s0 * (1.0f / 256.0f), mu1 = s1 * (1.0f / 256.0f);
        const float rs0 = rsqrtf(q0 * (1.0f / 256.0f) - mu0 * mu0 + LN_EPS);
        const float rs1 = rsqrtf(q1 * (1.0f / 256.0f) - mu1 * mu1 + LN_EPS);
#pragma unroll
        for (int i = 0; i < 8; ++i) {
            const int d = lane + 32 * i;
            union { _Float16 hh[2]; unsigned u; } pk;
            pk.hh[0] = (_Float16)((v0[i] - mu0) * rs0 * g[i] + h[i]);
            pk.hh[1] = (_Float16)((v1[i] - mu1) * rs1 * g[i] + h[i]);
            *reinterpret_cast<unsigned*>(&sT[d][cl]) = pk.u;   // b32 packed transpose store
        }
    }
    __syncthreads();

    // coalesced writeout: 8 consecutive threads cover 128B of one d-row
#pragma unroll
    for (int it = 0; it < 8; ++it) {
        const int d  = it * 32 + (threadIdx.x >> 3);
        const int ch = (threadIdx.x & 7) * 8;
        const v8h vv = *reinterpret_cast<const v8h*>(&sT[d][ch]);
        *reinterpret_cast<v8h*>(xembT + ((size_t)(b * 256 + d)) * 256 + c0 + ch) = vv;
    }
}

// ---------------- kE: out[b,p,d] = (sum_c mask[p,c]*x_emb[b,c,d]) * (1+ew[p]) + mask[b,p]*eb[b]
// grid (d-tiles=2, p-tiles=4, b=256); block 256 = 8 waves; wave: 1 p-subtile x 4 d-subtiles
#define KE_PITCH 40
__global__ void kE_gemm(const _Float16* __restrict__ maskh,
                        const _Float16* __restrict__ xembT,   // [b][d][c]
                        const float* __restrict__ maskf,
                        const float* __restrict__ expand_w,
                        const float* __restrict__ expand_b,
                        float* __restrict__ out) {
    __shared__ __align__(16) _Float16 sA[64][KE_PITCH];   // mask   [p_local][k]
    __shared__ __align__(16) _Float16 sB[128][KE_PITCH];  // xembT  [d_local][k]

    const int b  = blockIdx.z;
    const int p0 = blockIdx.y * 64;
    const int d0 = blockIdx.x * 128;
    const int tid  = threadIdx.x;
    const int wave = tid >> 5, lane = tid & 31;
    const int wp  = wave & 3;          // p subtile
    const int wdg = (wave >> 2) * 4;   // first of four d subtiles

    const int m  = lane & 15;
    const int kb = (lane >> 4) << 3;   // 0 or 8 (ISA 16-bit lane layout)

    v8f acc[4] = {{}, {}, {}, {}};

    const int arow = tid >> 2, ach = (tid & 3) * 8;            // sA staging map (64x4 chunks)
    const int brow0 = tid >> 2, bch0 = (tid & 3) * 8;          // sB chunk 0
    const int brow1 = (tid + 256) >> 2, bch1 = (tid & 3) * 8;  // sB chunk 1

    for (int c0 = 0; c0 < 256; c0 += 32) {
        __syncthreads();
        stage16(maskh + (size_t)(p0 + arow) * 256 + c0 + ach, &sA[arow][ach]);
        stage16(xembT + ((size_t)(b * 256 + d0 + brow0)) * 256 + c0 + bch0, &sB[brow0][bch0]);
        stage16(xembT + ((size_t)(b * 256 + d0 + brow1)) * 256 + c0 + bch1, &sB[brow1][bch1]);
        stage_wait();
        __syncthreads();

        const v8h alo = *reinterpret_cast<const v8h*>(&sA[wp * 16 + m][kb]);
        const v8h ahi = *reinterpret_cast<const v8h*>(&sA[wp * 16 + m][16 + kb]);
        const v16h af = __builtin_shufflevector(alo, ahi,
                          0, 1, 2, 3, 4, 5, 6, 7, 8, 9, 10, 11, 12, 13, 14, 15);
#pragma unroll
        for (int j = 0; j < 4; ++j) {
            const int dr = (wdg + j) * 16 + m;
            const v8h blo = *reinterpret_cast<const v8h*>(&sB[dr][kb]);
            const v8h bhi = *reinterpret_cast<const v8h*>(&sB[dr][16 + kb]);
            const v16h bf = __builtin_shufflevector(blo, bhi,
                              0, 1, 2, 3, 4, 5, 6, 7, 8, 9, 10, 11, 12, 13, 14, 15);
            acc[j] = __builtin_amdgcn_wmma_f32_16x16x32_f16(false, af, false, bf,
                                                            (short)0, acc[j], false, false);
        }
    }

    // epilogue: C/D layout: VGPR r -> row m = r + (lane>=16 ? 8 : 0), col n = lane&15
    const int n  = lane & 15;
    const int mh = (lane >> 4) * 8;
    const float ebb = expand_b[b];
#pragma unroll
    for (int r = 0; r < 8; ++r) {
        const int p = p0 + wp * 16 + mh + r;
        const float scale = 1.0f + expand_w[p];
        const float bias  = maskf[(size_t)b * 256 + p] * ebb;
        const size_t base = ((size_t)b * 256 + p) * 256 + d0;
#pragma unroll
        for (int j = 0; j < 4; ++j)
            out[base + (wdg + j) * 16 + n] = acc[j][r] * scale + bias;
    }
}

extern "C" void kernel_launch(void* const* d_in, const int* in_sizes, int n_in,
                              void* d_out, int out_size, void* d_ws, size_t ws_size,
                              hipStream_t stream) {
    const float* x            = (const float*)d_in[0];
    const float* prev_cell    = (const float*)d_in[1];
    const float* few          = (const float*)d_in[2];
    const float* feb          = (const float*)d_in[3];
    const float* ln_emb_w     = (const float*)d_in[4];
    const float* ln_emb_b     = (const float*)d_in[5];
    const float* ln_col_w     = (const float*)d_in[6];
    const float* ln_col_b     = (const float*)d_in[7];
    const float* ln_prompt_w  = (const float*)d_in[8];
    const float* ln_prompt_b  = (const float*)d_in[9];
    const float* dense_imp_w  = (const float*)d_in[10];
    const float* dense_imp_b  = (const float*)d_in[11];
    const float* emb_column   = (const float*)d_in[12];
    const float* emb_prompt   = (const float*)d_in[13];
    const float* expand_w     = (const float*)d_in[14];
    const float* expand_b     = (const float*)d_in[15];
    float* out = (float*)d_out;

    char* ws = (char*)d_ws;
    float*    x_prompt = (float*)(ws);               // 262144 B
    float*    x_col    = (float*)(ws + 262144);      // 262144 B
    float*    maskf    = (float*)(ws + 524288);      // 262144 B
    _Float16* maskh    = (_Float16*)(ws + 786432);   // 131072 B
    _Float16* xembT    = (_Float16*)(ws + 917504);   // 33554432 B, layout [b][d][c]

    kA_prompt<<<256, 256, 0, stream>>>(emb_prompt, prev_cell, ln_prompt_w, ln_prompt_b,
                                       dense_imp_w, dense_imp_b, x_prompt);
    kB_col<<<256, 256, 0, stream>>>(emb_column, ln_col_w, ln_col_b, x_col);
    kD_emb<<<1024, 256, 0, stream>>>(x, few, feb, ln_emb_w, ln_emb_b, xembT);
    kC_mask<<<256, 256, 0, stream>>>(x_prompt, x_col, maskf, maskh);
    kE_gemm<<<dim3(2, 4, 256), 256, 0, stream>>>(maskh, xembT, maskf,
                                                 expand_w, expand_b, out);
}
